// PathAttention_1563368095937
// MI455X (gfx1250) — matrix operand order repaired and checked by reading, not compiled, not run on hardware
//
#include <hip/hip_runtime.h>
#include <hip/hip_bf16.h>

typedef _Float16 h16;
typedef __attribute__((ext_vector_type(16))) _Float16 v16h;
typedef __attribute__((ext_vector_type(8)))  _Float16 v8h;
typedef __attribute__((ext_vector_type(8)))  float    v8f;

#define N_SEG   100000
#define C_DIM   256
#define H_HEAD  8
#define D_HEAD  32
#define K_WIN   128
#define M_TOK   131072
#define P_WIN   (M_TOK / K_WIN)   // 1024
#define QKV_C   (3 * C_DIM)       // 768

union F16x16 { v16h v; v8h h[2]; _Float16 e[16]; };

// ---- WMMA fragment builders (CDNA5 f16 16x16x32 layouts, ISA 7.12.2) ----
// A (16x32, MxK): lanes 0-15 row M=lane, K in [c0,c0+8) U [c0+16,c0+24), c0 = (lane<16)?0:8
__device__ __forceinline__ v16h afrag_h(const h16* row, int kbase, int lane) {
    int c0 = (lane & 16) ? 8 : 0;
    F16x16 f;
    f.h[0] = *(const v8h*)(row + kbase + c0);
    f.h[1] = *(const v8h*)(row + kbase + c0 + 16);
    return f.v;
}
// B (32x16, KxN): lane holds column n=lane%16; K in [c0,c0+8) U [c0+16,c0+24), c0 = (lane<16)?0:8
__device__ __forceinline__ v16h bfrag_h(const h16* row, int kbase, int lane) {
    int c0 = (lane & 16) ? 8 : 0;
    F16x16 f;
    f.h[0] = *(const v8h*)(row + kbase + c0);
    f.h[1] = *(const v8h*)(row + kbase + c0 + 16);
    return f.v;
}
// A fragment built from f32 source with on-the-fly f32->f16 convert
__device__ __forceinline__ v16h afrag_f32(const float* row, int kbase, int lane) {
    int c0 = (lane & 16) ? 8 : 0;
    F16x16 f;
#pragma unroll
    for (int i = 0; i < 8; ++i) {
        f.e[i]     = (_Float16)row[kbase + c0 + i];
        f.e[i + 8] = (_Float16)row[kbase + c0 + 16 + i];
    }
    return f.v;
}
__device__ __forceinline__ v8f wmma16(v16h a, v16h b, v8f c) {
    return __builtin_amdgcn_wmma_f32_16x16x32_f16(false, a, false, b, (short)0, c, false, false);
}
__device__ __forceinline__ v8f vzero8() { v8f z = {0.f,0.f,0.f,0.f,0.f,0.f,0.f,0.f}; return z; }

// ---------------- kernel 0: zero scratch (sums + counts) ----------------
__global__ void k_zero(float* p, long long n) {
    long long i = (long long)blockIdx.x * blockDim.x + threadIdx.x;
    if (i < n) p[i] = 0.f;
}

// ---------------- kernel 1: convert weights to f16 ----------------------
__global__ void k_cvtw(const float* wqkv, const float* wproj, h16* wqkv_h, h16* wproj_h) {
    int i = blockIdx.x * blockDim.x + threadIdx.x;
    if (i < QKV_C * C_DIM) wqkv_h[i] = (h16)wqkv[i];
    if (i < C_DIM * C_DIM) wproj_h[i] = (h16)wproj[i];
}

// ---------------- kernel 2: QKV GEMM  (N x 256) @ (768 x 256)^T ---------
// grid 3125 blocks x 384 thr (12 waves). Block tile: 32 rows x 768 cols.
// wave w owns cols [w*64, w*64+64), both 16-row m-tiles.
__global__ __launch_bounds__(384) void k_qkv(const float* __restrict__ feat,
                                             const h16* __restrict__ w_h,
                                             const float* __restrict__ bias,
                                             h16* __restrict__ qkv_h) {
    int lane = threadIdx.x & 31;
    int wave = threadIdx.x >> 5;            // 0..11
    int m_base = blockIdx.x * 32;
    int n0 = wave * 64;
    v8f acc[2][4];
#pragma unroll
    for (int a = 0; a < 2; ++a)
#pragma unroll
        for (int b = 0; b < 4; ++b) acc[a][b] = vzero8();

    const float* arow0 = feat + (long long)(m_base + (lane & 15)) * C_DIM;
    const float* arow1 = arow0 + 16 * C_DIM;
#pragma unroll
    for (int kk = 0; kk < C_DIM; kk += 32) {
        v16h a0 = afrag_f32(arow0, kk, lane);
        v16h a1 = afrag_f32(arow1, kk, lane);
#pragma unroll
        for (int nt = 0; nt < 4; ++nt) {
            const h16* brow = w_h + (long long)(n0 + nt * 16 + (lane & 15)) * C_DIM;
            v16h b = bfrag_h(brow, kk, lane);
            acc[0][nt] = wmma16(a0, b, acc[0][nt]);
            acc[1][nt] = wmma16(a1, b, acc[1][nt]);
        }
    }
    int coll = lane & 15;
    int rofs = (lane & 16) ? 8 : 0;
#pragma unroll
    for (int mt = 0; mt < 2; ++mt)
#pragma unroll
        for (int nt = 0; nt < 4; ++nt) {
            int col = n0 + nt * 16 + coll;
            float bv = bias[col];
#pragma unroll
            for (int r = 0; r < 8; ++r) {
                int row = m_base + mt * 16 + r + rofs;
                qkv_h[(long long)row * QKV_C + col] = (h16)(acc[mt][nt][r] + bv);
            }
        }
}

// ---------------- kernel 3: windowed attention --------------------------
// grid (1024, 8), 128 threads (4 waves). One (window p, head h) per block.
__global__ __launch_bounds__(128) void k_attn(const h16* __restrict__ qkv_h,
                                              const int* __restrict__ inv,
                                              const float* __restrict__ cosb,
                                              const float* __restrict__ sinb,
                                              h16* __restrict__ out_h) {
    __shared__ h16   q_s[K_WIN][D_HEAD + 8];
    __shared__ h16   k_s[K_WIN][D_HEAD + 8];
    __shared__ h16   vT_s[D_HEAD][K_WIN + 8];
    __shared__ float s_s[K_WIN][K_WIN + 4];
    __shared__ h16   p_s[K_WIN][K_WIN + 8];
    __shared__ float rsum_s[K_WIN];

    int p = blockIdx.x, h = blockIdx.y;
    int t = threadIdx.x;                    // token 0..127
    int mg = p * K_WIN + t;
    int row = inv[mg];
    const h16* base = qkv_h + (long long)row * QKV_C + h * D_HEAD;

    float q[D_HEAD], k[D_HEAD], v[D_HEAD];
#pragma unroll
    for (int d = 0; d < D_HEAD; ++d) {
        q[d] = (float)base[d];
        k[d] = (float)base[C_DIM + d];
        v[d] = (float)base[2 * C_DIM + d];
    }
    // RoPE (half-dim rotate, f32 math)
#pragma unroll
    for (int d = 0; d < D_HEAD / 2; ++d) {
        float c = cosb[(long long)mg * D_HEAD + d];
        float s = sinb[(long long)mg * D_HEAD + d];
        q_s[t][d]              = (h16)(q[d] * c - q[d + 16] * s);
        q_s[t][d + 16]         = (h16)(q[d + 16] * c + q[d] * s);
        k_s[t][d]              = (h16)(k[d] * c - k[d + 16] * s);
        k_s[t][d + 16]         = (h16)(k[d + 16] * c + k[d] * s);
    }
#pragma unroll
    for (int d = 0; d < D_HEAD; ++d) vT_s[d][t] = (h16)v[d];
    __syncthreads();

    int lane = threadIdx.x & 31, wave = threadIdx.x >> 5;

    // S = Q @ K^T : 8x8 tiles of 16x16, K-dim = 32 = one wmma step
#pragma unroll
    for (int mt = 0; mt < 2; ++mt) {
        int mtile = wave * 2 + mt;
        v16h a = afrag_h(&q_s[mtile * 16 + (lane & 15)][0], 0, lane);
#pragma unroll
        for (int nt = 0; nt < 8; ++nt) {
            v16h b = bfrag_h(&k_s[nt * 16 + (lane & 15)][0], 0, lane);
            v8f acc = vzero8();
            acc = wmma16(a, b, acc);
            int col = nt * 16 + (lane & 15);
            int ro = (lane & 16) ? 8 : 0;
#pragma unroll
            for (int r = 0; r < 8; ++r)
                s_s[mtile * 16 + r + ro][col] = acc[r];
        }
    }
    __syncthreads();

    // softmax over row t (normalization deferred to the PV epilogue)
    const float scale = 0.17677669529663687f;   // 1/sqrt(32)
    float mx = -1e30f;
    for (int j = 0; j < K_WIN; ++j) mx = fmaxf(mx, s_s[t][j]);
    mx *= scale;
    float sum = 0.f;
    for (int j = 0; j < K_WIN; ++j) {
        float e = __expf(s_s[t][j] * scale - mx);
        p_s[t][j] = (h16)e;
        sum += e;
    }
    rsum_s[t] = 1.0f / sum;
    __syncthreads();

    // O = P @ V : A = p_s (128x128), B = vT_s, 2 n-tiles of d
    v8f acc2[2][2];
#pragma unroll
    for (int a = 0; a < 2; ++a)
#pragma unroll
        for (int b = 0; b < 2; ++b) acc2[a][b] = vzero8();
#pragma unroll
    for (int kk = 0; kk < K_WIN; kk += 32) {
        v16h a0 = afrag_h(&p_s[(wave * 2 + 0) * 16 + (lane & 15)][0], kk, lane);
        v16h a1 = afrag_h(&p_s[(wave * 2 + 1) * 16 + (lane & 15)][0], kk, lane);
#pragma unroll
        for (int nt = 0; nt < 2; ++nt) {
            v16h b = bfrag_h(&vT_s[nt * 16 + (lane & 15)][0], kk, lane);
            acc2[0][nt] = wmma16(a0, b, acc2[0][nt]);
            acc2[1][nt] = wmma16(a1, b, acc2[1][nt]);
        }
    }
    int ro = (lane & 16) ? 8 : 0;
#pragma unroll
    for (int mt = 0; mt < 2; ++mt)
#pragma unroll
        for (int nt = 0; nt < 2; ++nt) {
            int d = nt * 16 + (lane & 15);
#pragma unroll
            for (int r = 0; r < 8; ++r) {
                int i = (wave * 2 + mt) * 16 + r + ro;
                out_h[((long long)(p * K_WIN + i)) * C_DIM + h * D_HEAD + d] =
                    (h16)(acc2[mt][nt][r] * rsum_s[i]);
            }
        }
}

// ---------------- kernel 4: segment counts ------------------------------
__global__ void k_cnt(const int* __restrict__ inv, float* __restrict__ cnt) {
    int m = blockIdx.x * 256 + threadIdx.x;
    if (m < M_TOK) atomicAdd(&cnt[inv[m]], 1.0f);
}

// ---------------- kernel 5: proj GEMM fused with segment-sum scatter ----
// grid 4096 x 256 thr (8 waves). Block tile 32 rows x 256 cols.
__global__ __launch_bounds__(256) void k_proj(const h16* __restrict__ attn_h,
                                              const h16* __restrict__ wproj_h,
                                              const float* __restrict__ bias,
                                              const int* __restrict__ inv,
                                              float* __restrict__ sums) {
    int lane = threadIdx.x & 31;
    int wv = threadIdx.x >> 5;              // 0..7
    int ng = wv & 3;                        // 4 col groups * 64
    int mt = wv >> 2;                       // 0..1
    long long m_base = (long long)blockIdx.x * 32 + mt * 16;
    int n0 = ng * 64;

    v8f acc[4];
#pragma unroll
    for (int a = 0; a < 4; ++a) acc[a] = vzero8();

    const h16* arow = attn_h + (m_base + (lane & 15)) * C_DIM;
#pragma unroll
    for (int kk = 0; kk < C_DIM; kk += 32) {
        v16h a = afrag_h(arow, kk, lane);
#pragma unroll
        for (int nt = 0; nt < 4; ++nt) {
            const h16* brow = wproj_h + (long long)(n0 + nt * 16 + (lane & 15)) * C_DIM;
            v16h b = bfrag_h(brow, kk, lane);
            acc[nt] = wmma16(a, b, acc[nt]);
        }
    }
    int ro = (lane & 16) ? 8 : 0;
#pragma unroll
    for (int nt = 0; nt < 4; ++nt) {
        int col = n0 + nt * 16 + (lane & 15);
        float bv = bias[col];
#pragma unroll
        for (int r = 0; r < 8; ++r) {
            long long row = m_base + r + ro;
            int seg = inv[row];
            atomicAdd(&sums[(long long)seg * C_DIM + col], acc[nt][r] + bv);
        }
    }
}

// ---------------- kernel 6: finalize mean -------------------------------
__global__ void k_final(const float* __restrict__ sums, const float* __restrict__ cnt,
                        float* __restrict__ out, long long total) {
    long long i = (long long)blockIdx.x * 256 + threadIdx.x;
    if (i < total) {
        float c = cnt[i >> 8];
        out[i] = (c > 0.f) ? sums[i] / c : 0.f;
    }
}

extern "C" void kernel_launch(void* const* d_in, const int* in_sizes, int n_in,
                              void* d_out, int out_size, void* d_ws, size_t ws_size,
                              hipStream_t stream) {
    const float* feat  = (const float*)d_in[0];
    const int*   inv   = (const int*)d_in[2];
    const float* cosb  = (const float*)d_in[4];
    const float* sinb  = (const float*)d_in[5];
    const float* wqkv  = (const float*)d_in[6];
    const float* bqkv  = (const float*)d_in[7];
    const float* wproj = (const float*)d_in[8];
    const float* bproj = (const float*)d_in[9];
    float* out = (float*)d_out;

    // workspace layout (all 256B aligned)
    char* ws = (char*)d_ws;
    size_t o = 0;
    h16* qkv_h  = (h16*)(ws + o); o += (size_t)N_SEG * QKV_C * sizeof(h16);      // 153.6 MB
    h16* attn_h = (h16*)(ws + o); o += (size_t)M_TOK * C_DIM * sizeof(h16);      //  67.1 MB
    h16* wqkv_h = (h16*)(ws + o); o += (size_t)QKV_C * C_DIM * sizeof(h16);
    o = (o + 255) & ~(size_t)255;
    h16* wproj_h = (h16*)(ws + o); o += (size_t)C_DIM * C_DIM * sizeof(h16);
    o = (o + 255) & ~(size_t)255;
    float* sums = (float*)(ws + o); o += (size_t)N_SEG * C_DIM * sizeof(float);  // 102.4 MB
    float* cnt  = (float*)(ws + o); o += (size_t)N_SEG * sizeof(float);
    (void)ws_size; (void)in_sizes; (void)n_in; (void)out_size;

    // 0) zero sums + counts (contiguous)
    long long nz = (long long)N_SEG * C_DIM + N_SEG;
    k_zero<<<(int)((nz + 255) / 256), 256, 0, stream>>>(sums, nz);
    // 1) weights -> f16
    k_cvtw<<<(QKV_C * C_DIM + 255) / 256, 256, 0, stream>>>(wqkv, wproj, wqkv_h, wproj_h);
    // 2) QKV GEMM
    k_qkv<<<N_SEG / 32, 384, 0, stream>>>(feat, wqkv_h, bqkv, qkv_h);
    // 3) attention per (window, head)
    k_attn<<<dim3(P_WIN, H_HEAD), 128, 0, stream>>>(qkv_h, inv, cosb, sinb, attn_h);
    // 4) segment counts
    k_cnt<<<M_TOK / 256, 256, 0, stream>>>(inv, cnt);
    // 5) proj GEMM + segment-sum scatter
    k_proj<<<M_TOK / 32, 256, 0, stream>>>(attn_h, wproj_h, bproj, inv, sums);
    // 6) finalize mean
    long long tot = (long long)N_SEG * C_DIM;
    k_final<<<(int)((tot + 255) / 256), 256, 0, stream>>>(sums, cnt, out, tot);
}
